// FlumenHead_88287347736654
// MI455X (gfx1250) — matrix operand order repaired and checked by reading, not compiled, not run on hardware
//
#include <hip/hip_runtime.h>
#include <hip/hip_bf16.h>
#include <math.h>

// ---------------------------------------------------------------------------
// FlumenHead: MLP encoder + 65536-step LSTM scan (hidden=256, in=32).
// Latency-bound recurrence -> persistent kernel, weights resident in LDS,
// h split across 16 co-resident workgroups, per-step L2 exchange + atomic
// barrier, recurrent GEMV slice done with v_wmma_f32_16x16x32_f16.
// Input u[t+1] is register-prefetched one step ahead so its global-load
// latency overlaps the WMMA chain + inter-WG barrier of step t.
// ---------------------------------------------------------------------------

typedef __attribute__((ext_vector_type(16))) _Float16 v16h;
typedef __attribute__((ext_vector_type(8)))  _Float16 v8h;
typedef __attribute__((ext_vector_type(8)))  float    v8f;

#define HID   256
#define INDIM 32
#define SEQ   65536
#define NWG   16          // workgroups cooperating on the recurrence
#define HPW   16          // h elements owned per workgroup
#define ROWS  64          // 4 gates * HPW z-rows per workgroup
#define KCH   8           // 256 / 32 K-chunks per WMMA accumulation chain
#define WPAD  (HID + 8)   // LDS row pad: stride 528B -> 16 rows hit distinct banks

__device__ __forceinline__ float sigmoidf_(float x) {
  return 1.0f / (1.0f + __expf(-x));
}

// ------------------------- encoder: h0 = MLP(s0) ---------------------------
__global__ __launch_bounds__(256) void flumen_encoder(
    const float* __restrict__ s, const float* __restrict__ W1,
    const float* __restrict__ b1, const float* __restrict__ W2,
    const float* __restrict__ b2, const float* __restrict__ W3,
    const float* __restrict__ b3, float* __restrict__ hbuf0) {
  __shared__ float s0[64];
  __shared__ float x1[256];
  __shared__ float x2[256];
  const int tid = threadIdx.x;
  if (tid < 64) s0[tid] = s[tid];
  __syncthreads();
  float a = b1[tid];
  #pragma unroll 4
  for (int k = 0; k < 64; ++k) a += W1[tid * 64 + k] * s0[k];
  x1[tid] = tanhf(a);
  __syncthreads();
  a = b2[tid];
  #pragma unroll 4
  for (int k = 0; k < 256; ++k) a += W2[tid * 256 + k] * x1[k];
  x2[tid] = tanhf(a);
  __syncthreads();
  a = b3[tid];
  #pragma unroll 4
  for (int k = 0; k < 256; ++k) a += W3[tid * 256 + k] * x2[k];
  hbuf0[tid] = a;
}

// --------------------- persistent LSTM scan kernel -------------------------
__global__ __launch_bounds__(256) void flumen_lstm(
    const float* __restrict__ in,    // [SEQ][INDIM]
    const float* __restrict__ Wih,   // [1024][INDIM]
    const float* __restrict__ Whh,   // [1024][HID]
    const float* __restrict__ bias,  // [1024]
    float* __restrict__ hbuf,        // ws: [2][HID] fp32 h exchange
    unsigned* __restrict__ ctr,      // ws: barrier counter (memset to 0)
    float* __restrict__ out) {       // [HID h_last][SEQ*HID h_seq]
  const int g    = blockIdx.x;       // 0..NWG-1
  const int tid  = threadIdx.x;
  const int lane = tid & 31;
  const int wave = tid >> 5;

  __shared__ _Float16 sWhh[ROWS][WPAD];     // f16 recurrent weight slice
  __shared__ float    sWih[ROWS][INDIM + 1];
  __shared__ float    sB[ROWS];
  __shared__ _Float16 sHf[HID];             // current h, f16 (WMMA operand)
  __shared__ float    sU[INDIM];
  __shared__ float    sZ[4][HPW];           // WMMA z-tiles per gate
  __shared__ float    sZin[ROWS];           // input-GEMV contribution

  // ---- one-time staging: this WG's 64 z-rows (gate G, h-index g*16+m) ----
  for (int i = tid; i < ROWS * HID; i += 256) {
    int r = i / HID, k = i - r * HID;
    int grow = (r >> 4) * HID + g * HPW + (r & 15);
    sWhh[r][k] = (_Float16)Whh[grow * HID + k];
  }
  for (int i = tid; i < ROWS * INDIM; i += 256) {
    int r = i / INDIM, k = i - r * INDIM;
    int grow = (r >> 4) * HID + g * HPW + (r & 15);
    sWih[r][k] = Wih[grow * INDIM + k];
  }
  if (tid < ROWS) {
    int grow = (tid >> 4) * HID + g * HPW + (tid & 15);
    sB[tid] = bias[grow];
  }
  if (tid < HID) sHf[tid] = (_Float16)hbuf[tid];   // h0 from encoder
  float hOwn = 0.0f, cOwn = 0.0f;                  // owned slice of h, c
  if (tid < HPW) hOwn = hbuf[g * HPW + tid];
  float uReg = 0.0f;                               // prefetched u[t]
  if (tid < INDIM) uReg = in[tid];
  __syncthreads();

  float* __restrict__ seq_out = out + HID;
  // A-operand K striping (ISA 16-bit A 16x32 table): lanes 0-15 hold
  // K = {0..7,16..23}, lanes 16-31 hold K = {8..15,24..31}.
  const int aoff = (lane & 16) ? 8 : 0;
  const int mrow = lane & 15;
  // B-operand (32x16, columns all = h chunk): lanes 0-15 hold
  // K = {0..7,16..23}, lanes 16-31 hold K = {8..15,24..31}, 2 K-values per VGPR.
  const int boff = (lane & 16) ? 8 : 0;

  for (int t = 0; t < SEQ; ++t) {
    if (tid < INDIM) sU[tid] = uReg;   // u[t], prefetched last iteration
    if (tid < HPW)   // h_seq collects the PRE-step hidden state
      seq_out[(size_t)t * HID + g * HPW + tid] = hOwn;
    __syncthreads();

    // Prefetch u[t+1]: load issues now, waitcnt lands at next iteration's
    // sU store, fully hidden behind the WMMA chain + inter-WG barrier.
    if (tid < INDIM && t + 1 < SEQ) uReg = in[(t + 1) * INDIM + tid];

    if (wave < 4) {
      // z-tile for gate `wave`: D = sum_c A(16x32) x B(32x16), B columns
      // all equal h-chunk, so every D column equals the 16-row z slice.
      v8f acc = {};
      const int rbase = wave * HPW + mrow;
      #pragma unroll
      for (int c = 0; c < KCH; ++c) {
        union { v16h v; v8h h[2]; } a, bm;
        a.h[0]  = *(const v8h*)&sWhh[rbase][c * 32 + aoff];
        a.h[1]  = *(const v8h*)&sWhh[rbase][c * 32 + aoff + 16];
        bm.h[0] = *(const v8h*)&sHf[c * 32 + boff];
        bm.h[1] = *(const v8h*)&sHf[c * 32 + boff + 16];
        acc = __builtin_amdgcn_wmma_f32_16x16x32_f16(
            false, a.v, false, bm.v, (short)0, acc, false, false);
      }
      if (mrow == 0) {  // column N=0: VGPR r holds M=r (lanes 0-15) / r+8
        const int mb = (lane & 16) ? 8 : 0;
        #pragma unroll
        for (int r = 0; r < 8; ++r) sZ[wave][mb + r] = acc[r];
      }
    } else if (wave < 6) {
      // input contribution: 64 rows x 32 MACs
      const int r = tid - 128;
      float a = 0.0f;
      #pragma unroll
      for (int k = 0; k < INDIM; ++k) a += sWih[r][k] * sU[k];
      sZin[r] = a;
    }
    __syncthreads();

    if (tid < HPW) {
      float zi = sZ[0][tid] + sZin[tid]      + sB[tid];
      float zf = sZ[1][tid] + sZin[16 + tid] + sB[16 + tid];
      float zg = sZ[2][tid] + sZin[32 + tid] + sB[32 + tid];
      float zo = sZ[3][tid] + sZin[48 + tid] + sB[48 + tid];
      cOwn = sigmoidf_(zf) * cOwn + sigmoidf_(zi) * tanhf(zg);
      hOwn = sigmoidf_(zo) * tanhf(cOwn);
      // agent-scope store: bypasses WGP cache, lands in device-coherent L2
      __hip_atomic_store(&hbuf[((t + 1) & 1) * HID + g * HPW + tid], hOwn,
                         __ATOMIC_RELAXED, __HIP_MEMORY_SCOPE_AGENT);
      __threadfence();  // drain storecnt before signalling the barrier
    }
    __syncthreads();

    // inter-WG barrier (monotone counter; double-buffered hbuf makes one
    // barrier per step sufficient)
    if (tid == 0) {
      __hip_atomic_fetch_add(ctr, 1u, __ATOMIC_ACQ_REL,
                             __HIP_MEMORY_SCOPE_AGENT);
      const unsigned target = (unsigned)(t + 1) * NWG;
      while (__hip_atomic_load(ctr, __ATOMIC_ACQUIRE,
                               __HIP_MEMORY_SCOPE_AGENT) < target)
        __builtin_amdgcn_s_sleep(1);
    }
    __syncthreads();

    if (tid < HID) {
      float hv = __hip_atomic_load(&hbuf[((t + 1) & 1) * HID + tid],
                                   __ATOMIC_RELAXED, __HIP_MEMORY_SCOPE_AGENT);
      sHf[tid] = (_Float16)hv;
    }
    __syncthreads();
  }

  if (tid < HPW) out[g * HPW + tid] = hOwn;  // h_last
}

// ---------------------------------------------------------------------------
extern "C" void kernel_launch(void* const* d_in, const int* in_sizes, int n_in,
                              void* d_out, int out_size, void* d_ws,
                              size_t ws_size, hipStream_t stream) {
  const float* s0  = (const float*)d_in[0];
  const float* rin = (const float*)d_in[1];
  const float* Wih = (const float*)d_in[2];
  const float* Whh = (const float*)d_in[3];
  const float* b   = (const float*)d_in[4];
  const float* W1  = (const float*)d_in[5];
  const float* b1  = (const float*)d_in[6];
  const float* W2  = (const float*)d_in[7];
  const float* b2  = (const float*)d_in[8];
  const float* W3  = (const float*)d_in[9];
  const float* b3  = (const float*)d_in[10];
  float* out = (float*)d_out;

  // workspace layout: [2][256] fp32 h exchange buffers, then barrier counter
  float*    hbuf = (float*)d_ws;
  unsigned* ctr  = (unsigned*)((char*)d_ws + 2 * HID * sizeof(float));

  hipMemsetAsync(ctr, 0, sizeof(unsigned), stream);
  flumen_encoder<<<1, 256, 0, stream>>>(s0, W1, b1, W2, b2, W3, b3, hbuf);
  flumen_lstm<<<NWG, 256, 0, stream>>>(rin, Wih, Whh, b, hbuf, ctr, out);
}
